// GCNAggregator_33767032881499
// MI455X (gfx1250) — matrix operand order repaired and checked by reading, not compiled, not run on hardware
//
#include <hip/hip_runtime.h>

// ---- CDNA5 (gfx1250) wave32 WMMA types ----
typedef __bf16 bf16;
typedef __attribute__((ext_vector_type(8)))  __bf16 v8bf;
typedef __attribute__((ext_vector_type(16))) __bf16 v16bf;
typedef __attribute__((ext_vector_type(8)))  float  v8f;
typedef __attribute__((ext_vector_type(4)))  float  v4f;

#define NNODES 50000
#define DEG    16
#define F      128          // F_IN == F_OUT
#define NPB    16           // nodes per block (16 neighbor tiles + 1 agg tile)

// Fused GCN aggregate+transform. Memory-bound workload (~870 MB f32 traffic,
// ~37 us floor @ 23.3 TB/s; 28 GFLOP). Each block: 16 nodes = 16 neighbor
// tiles (16x128) streamed once through LDS (f32->bf16), column sums fused in
// for the segment mean; 8 wave32s each hold a 128x16 bf16 W.T slice in
// registers and run v_wmma_f32_16x16x32_bf16 (K=128 -> 4 WMMA per tile).
__global__ __launch_bounds__(256) void gcn_fused_wmma(
    const float* __restrict__ x,
    const float* __restrict__ neigh,
    const float* __restrict__ W,       // [F_OUT][F_IN] row-major
    float* __restrict__ x_out,         // [N][F]
    float* __restrict__ neigh_out)     // [E][F]
{
    __shared__ bf16  abuf[2][16][F];     // double-buffered A tiles (bf16)
    __shared__ bf16  aggbuf[16][F];      // x + mean(neigh) tile (bf16)
    __shared__ float psum[2][2][F];      // partial column sums (double-buffered)

    const int tid  = threadIdx.x;
    const int lane = tid & 31;           // wave32 lane
    const int w    = tid >> 5;           // wave id 0..7 -> output cols 16w..16w+15
    const int hi   = lane >> 4;          // half-wave select
    const int ln   = lane & 15;
    const int nb   = blockIdx.x * NPB;   // first node of this group

    // ---- B fragments: W.T slice for this wave's 16 output columns ----
    // B operand, k order chosen here (same as A): lanes 0-15 hold K=0..15 of column `ln`,
    // lanes 16-31 hold K=16..31.  B[k][n] = W[n][k] -> contiguous along W row.
    // W is small & shared by all blocks: regular (cached) loads.
    v16bf bfrag[4];
    {
        const float* wrow = W + (size_t)(16 * w + ln) * F + hi * 16;
        #pragma unroll
        for (int kb = 0; kb < 4; ++kb) {
            const v4f* p4 = (const v4f*)(wrow + kb * 32);   // 64B-aligned
            v16bf b;
            #pragma unroll
            for (int q = 0; q < 4; ++q) {
                v4f f = p4[q];
                #pragma unroll
                for (int j = 0; j < 4; ++j) b[4 * q + j] = (bf16)f[j];
            }
            bfrag[kb] = b;
        }
    }

    // ---- loader mapping: thread covers column c, rows half*8..half*8+7 ----
    const int c    = tid & 127;
    const int half = tid >> 7;

    for (int t = 0; t < NPB + 1; ++t) {
        // ---------- load tile t (t<16): global f32 -> LDS bf16, + column sums
        if (t < NPB) {
            const int buf = t & 1;
            const float* src =
                neigh + ((size_t)(nb + t) * DEG + half * 8) * F + c;
            float s = 0.0f;
            #pragma unroll
            for (int i = 0; i < 8; ++i) {
                float v = __builtin_nontemporal_load(src + (size_t)i * F);
                s += v;
                abuf[buf][half * 8 + i][c] = (bf16)v;
            }
            psum[buf][half][c] = s;
        }
        __syncthreads();

        // ---------- finish agg row t: x[nb+t] + mean of its 16 neighbors
        if (t < NPB && tid < 128) {
            const int buf = t & 1;
            float s = psum[buf][0][c] + psum[buf][1][c];
            float a = __builtin_nontemporal_load(x + (size_t)(nb + t) * F + c)
                      + s * (1.0f / 16.0f);
            aggbuf[t][c] = (bf16)a;
        }

        // ---------- WMMA: tile t (neighbors) or tile 16 (agg) ----------
        const bf16* asrc = (t < NPB) ? &abuf[t & 1][0][0] : &aggbuf[0][0];

        // Batch all 8 ds_load_b128 first so the scheduler can overlap LDS
        // latency with the WMMA chain (partial s_wait_dscnt, not 4 full drains).
        v8bf alo[4], ahi8[4];
        #pragma unroll
        for (int kb = 0; kb < 4; ++kb) {
            // A operand, k order chosen here (same as B): lane<16 -> K{0..15},
            // lane>=16 -> K{16..31}; row = ln.
            const bf16* base = asrc + (size_t)ln * F + kb * 32 + hi * 16;
            alo[kb]  = *(const v8bf*)(base);
            ahi8[kb] = *(const v8bf*)(base + 8);
        }

        v8f acc = {};
        #pragma unroll
        for (int kb = 0; kb < 4; ++kb) {
            v16bf afrag = __builtin_shufflevector(
                alo[kb], ahi8[kb],
                0, 1, 2, 3, 4, 5, 6, 7, 8, 9, 10, 11, 12, 13, 14, 15);
            acc = __builtin_amdgcn_wmma_f32_16x16x32_bf16(
                false, afrag, false, bfrag[kb], (short)0, acc, false, false);
        }

        // ---------- store: C layout VGPR v -> row (v + 8*hi), col = 16w+ln
        const int col = 16 * w + ln;
        float* outp;
        size_t rbase;
        if (t < NPB) { outp = neigh_out; rbase = (size_t)(nb + t) * DEG; }
        else         { outp = x_out;     rbase = (size_t)nb; }
        #pragma unroll
        for (int v = 0; v < 8; ++v)
            __builtin_nontemporal_store(
                acc[v], &outp[(rbase + v + 8 * hi) * F + col]);
    }
}

extern "C" void kernel_launch(void* const* d_in, const int* in_sizes, int n_in,
                              void* d_out, int out_size, void* d_ws, size_t ws_size,
                              hipStream_t stream) {
    // setup_inputs order: x[N*128] f32, neigh[E*128] f32, seg_ids[E] i32, W[128*128] f32
    const float* x     = (const float*)d_in[0];
    const float* neigh = (const float*)d_in[1];
    // d_in[2] = seg_ids: reference guarantees seg_ids[e] == e / DEG (sorted,
    // exactly DEG neighbors per node) -> structure is exploited directly.
    const float* W     = (const float*)d_in[3];

    float* x_out     = (float*)d_out;                          // [N][128]
    float* neigh_out = (float*)d_out + (size_t)NNODES * F;     // [E][128]

    dim3 grid(NNODES / NPB);   // 3125 groups of 16 nodes
    dim3 block(256);           // 8 wave32
    gcn_fused_wmma<<<grid, block, 0, stream>>>(x, neigh, W, x_out, neigh_out);
}
